// Correlation2D_78932908966238
// MI455X (gfx1250) — matrix operand order repaired and checked by reading, not compiled, not run on hardware
//
#include <hip/hip_runtime.h>
#include <math.h>

// ---------------------------------------------------------------------------
// RAFT correlation volume, fused for MI455X (gfx1250, wave32, WMMA).
//   cv = (f1^T f2)/16  ->  4-level avgpool pyramid  ->  81-pt bilinear lookup
// One workgroup (8 wave32) per 16 query pixels. The 16x8192 correlation slab
// is produced tile-by-tile with v_wmma_f32_16x16x32_bf16 and reduced on the
// fly. B-fragments are register double-buffered with a one-tile prefetch
// distance (amdgpu_waves_per_eu(2) raises the VGPR budget to 512 so the
// in-flight fragments are NOT re-sunk next to the WMMAs).
//  - level-1 built from accumulators via __shfl_xor lane-pair pooling
//  - level-0 kept only as per-query 10x10 windows (integer sample offsets)
//  - level-2/3 pooled from LDS-resident level-1
// HBM traffic ~27MB total; fmap2 (bf16, 4MB) stays L2-resident.
// ---------------------------------------------------------------------------

typedef __attribute__((ext_vector_type(16))) __bf16 v16bf;
typedef __attribute__((ext_vector_type(8)))  __bf16 v8bf;
typedef __attribute__((ext_vector_type(8)))  float  v8f;

#define Hf   64
#define Wf   128
#define HW   8192            // Hf*Wf
#define CH   256
#define QPB  16              // queries per block == WMMA M
#define TPB  256             // 8 waves of 32
#define NTILES (HW / QPB)    // 512 workgroups

// ---- prep: transpose [256,8192] f32 -> [8192,256] bf16 (1/16 folded into A)
__global__ __launch_bounds__(TPB) void corr_prep_kernel(
    const float* __restrict__ f1, const float* __restrict__ f2,
    __bf16* __restrict__ At, __bf16* __restrict__ Bt) {
  const int NE = HW * CH;
  int o = blockIdx.x * TPB + threadIdx.x;
  if (o < NE) {
    int m = o >> 8, k = o & 255;
    At[o] = (__bf16)(f1[k * HW + m] * 0.0625f);   // 1/sqrt(256)
  } else {
    int p = o - NE;
    int m = p >> 8, k = p & 255;
    Bt[p] = (__bf16)(f2[k * HW + m]);
  }
}

// ---- generic bilinear sample from an LDS-resident level map (zeros pad) ----
__device__ __forceinline__ float bilin_lds(const __bf16* base, int Hl, int Wl,
                                           float x, float y) {
  float x0f = floorf(x), y0f = floorf(y);
  int x0 = (int)x0f, y0 = (int)y0f;
  int x1 = x0 + 1, y1 = y0 + 1;
  float fx = x - x0f, fy = y - y0f;
  int x0c = x0 < 0 ? 0 : (x0 > Wl - 1 ? Wl - 1 : x0);
  int x1c = x1 < 0 ? 0 : (x1 > Wl - 1 ? Wl - 1 : x1);
  int y0c = y0 < 0 ? 0 : (y0 > Hl - 1 ? Hl - 1 : y0);
  int y1c = y1 < 0 ? 0 : (y1 > Hl - 1 ? Hl - 1 : y1);
  float mx0 = (x0 >= 0 && x0 < Wl) ? 1.f : 0.f;
  float mx1 = (x1 >= 0 && x1 < Wl) ? 1.f : 0.f;
  float my0 = (y0 >= 0 && y0 < Hl) ? 1.f : 0.f;
  float my1 = (y1 >= 0 && y1 < Hl) ? 1.f : 0.f;
  float v00 = (float)base[y0c * Wl + x0c] * (mx0 * my0);
  float v01 = (float)base[y0c * Wl + x1c] * (mx1 * my0);
  float v10 = (float)base[y1c * Wl + x0c] * (mx0 * my1);
  float v11 = (float)base[y1c * Wl + x1c] * (mx1 * my1);
  return (1.f - fy) * ((1.f - fx) * v00 + fx * v01) +
         fy * ((1.f - fx) * v10 + fx * v11);
}

__device__ __forceinline__ void sample_level(const __bf16* buf, int lvl, int Hl,
                                             int Wl, const float* coords,
                                             int tile, float* out, int tid) {
  const float inv = 1.0f / (float)(1 << lvl);
  for (int i = tid; i < QPB * 81; i += TPB) {
    int q = i / 81, o = i % 81;
    int iy = o / 9, ix = o % 9;
    int qg = tile * QPB + q;
    float x = coords[qg] * inv + (float)(ix - 4);
    float y = coords[HW + qg] * inv + (float)(iy - 4);
    out[(lvl * 81 + o) * HW + qg] = bilin_lds(buf + q * Hl * Wl, Hl, Wl, x, y);
  }
}

__device__ __forceinline__ void pool_level(const __bf16* src, __bf16* dst,
                                           int Hd, int Wd, int tid) {
  const int per = Hd * Wd;
  const int Ws = Wd * 2;
  for (int i = tid; i < QPB * per; i += TPB) {
    int q = i / per, r = i % per;
    int h = r / Wd, w = r % Wd;
    const __bf16* s = src + q * (per * 4) + (2 * h) * Ws + 2 * w;
    float sum = (float)s[0] + (float)s[1] + (float)s[Ws] + (float)s[Ws + 1];
    dst[i] = (__bf16)(0.25f * sum);
  }
}

// ---- fused correlation + pyramid + sampling --------------------------------
__global__ __launch_bounds__(TPB)
__attribute__((amdgpu_waves_per_eu(2)))   // allow ~512 VGPRs: keep prefetch live
void corr_main_kernel(
    const __bf16* __restrict__ At, const __bf16* __restrict__ Bt,
    const float* __restrict__ coords, float* __restrict__ out) {
  __shared__ __bf16 s_l1[QPB * 32 * 64];   // 64 KB level-1 (full)
  __shared__ __bf16 s_l2[QPB * 16 * 32];   // 16 KB level-2 (full)
  __shared__ __bf16 s_l3[QPB * 8 * 16];    //  4 KB level-3 (full)
  __shared__ __bf16 s_win[QPB * 10 * 10];  // level-0 sampling windows
  __shared__ int    s_wx0[QPB];
  __shared__ int    s_wy0[QPB];

  const int tid  = threadIdx.x;
  const int tile = blockIdx.x;
  const int wave = tid >> 5;
  const int lane = tid & 31;

  if (tid < QPB) {
    float cx = coords[tile * QPB + tid];
    float cy = coords[HW + tile * QPB + tid];
    s_wx0[tid] = (int)floorf(cx) - 4;
    s_wy0[tid] = (int)floorf(cy) - 4;
  }
  for (int i = tid; i < QPB * 100; i += TPB) s_win[i] = (__bf16)0.0f;
  __syncthreads();

  // Preload A fragments (16 query rows, full K=256) in contiguous k order, matching B:
  // lanes 0-15 row M=lane hold K = ks*32+0..15 (contiguous);
  // lanes 16-31 row M=lane-16 hold K = ks*32+16..31.
  v16bf afrag[8];
  {
    const int arow = tile * QPB + (lane & 15);
    const __bf16* Ap = At + arow * CH + ((lane >> 4) * 16);
#pragma unroll
    for (int ks = 0; ks < 8; ++ks) {
      v8bf lo = *(const v8bf*)(Ap + ks * 32);
      v8bf hi = *(const v8bf*)(Ap + ks * 32 + 8);
      v16bf a;
#pragma unroll
      for (int e = 0; e < 8; ++e) { a[e] = lo[e]; a[e + 8] = hi[e]; }
      afrag[ks] = a;
    }
  }

  // Hoist this lane's 8 query-window origins (q = mb..mb+7) into registers.
  const int c16 = lane & 15;
  const int mb  = (lane >> 4) * 8;   // C/D layout: lanes 0-15 M=v, 16-31 M=v+8
  int wx0r[8], wy0r[8];
#pragma unroll
  for (int v = 0; v < 8; ++v) { wx0r[v] = s_wx0[mb + v]; wy0r[v] = s_wy0[mb + v]; }

  // Wave w owns feature-map rows h2 in [8w, 8w+8): 4 row-pairs x 8 col-blocks.
  // B layout: lanes 0-15 col N=lane hold K = ks*32+0..15 (contiguous),
  // lanes 16-31 col N=lane-16 hold K = ks*32+16..31.
  for (int p = 0; p < 4; ++p) {
    const int r0 = wave * 8 + p * 2;
    const __bf16* Bbase = Bt + (r0 * Wf + c16) * CH + ((lane >> 4) * 16);

    // Register double-buffer: prefetch tile j+1 while running tile j's WMMAs.
    v16bf b0buf[2][8], b1buf[2][8];
#pragma unroll
    for (int ks = 0; ks < 8; ++ks) {          // prime j = 0
      b0buf[0][ks] = *(const v16bf*)(Bbase + ks * 32);
      b1buf[0][ks] = *(const v16bf*)(Bbase + Wf * CH + ks * 32);
    }

#pragma unroll
    for (int j = 0; j < 8; ++j) {
      const int cb = j & 1;
      if (j + 1 < 8) {                        // constant-folded (full unroll)
        const __bf16* Bn = Bbase + (j + 1) * 16 * CH;
#pragma unroll
        for (int ks = 0; ks < 8; ++ks) {
          b0buf[cb ^ 1][ks] = *(const v16bf*)(Bn + ks * 32);
          b1buf[cb ^ 1][ks] = *(const v16bf*)(Bn + Wf * CH + ks * 32);
        }
      }
      // two independent 8-deep WMMA accumulate chains (co-executable)
      v8f acc0 = {};
      v8f acc1 = {};
#pragma unroll
      for (int ks = 0; ks < 8; ++ks) {
        acc0 = __builtin_amdgcn_wmma_f32_16x16x32_bf16(
            false, afrag[ks], false, b0buf[cb][ks], (short)0, acc0, false, false);
        acc1 = __builtin_amdgcn_wmma_f32_16x16x32_bf16(
            false, afrag[ks], false, b1buf[cb][ks], (short)0, acc1, false, false);
      }
      // reduce straight from registers:
      //   2x2 avg-pool: rows in-lane (acc0+acc1), columns via lane-pair shfl
      //   L0 window capture: predicated ds_store_b16 per element
      const int gx = j * 16 + c16;
#pragma unroll
      for (int v = 0; v < 8; ++v) {
        const int q = mb + v;
        float ps = acc0[v] + acc1[v];
        ps += __shfl_xor(ps, 1, 32);
        if ((lane & 1) == 0)
          s_l1[q * (32 * 64) + (r0 >> 1) * 64 + j * 8 + (c16 >> 1)] =
              (__bf16)(0.25f * ps);
        const int dx  = gx - wx0r[v];
        const int dy0 = r0 - wy0r[v];
        if ((unsigned)dx < 10u) {
          if ((unsigned)dy0 < 10u)
            s_win[(q * 10 + dy0) * 10 + dx] = (__bf16)acc0[v];
          if ((unsigned)(dy0 + 1) < 10u)
            s_win[(q * 10 + dy0 + 1) * 10 + dx] = (__bf16)acc1[v];
        }
      }
    }
  }
  __syncthreads();

  // B1: level-0 output from windows (integer offsets -> shared fractions);
  //     pool L1 -> L2.
  for (int i = tid; i < QPB * 81; i += TPB) {
    int q = i / 81, o = i % 81;
    int iy = o / 9, ix = o % 9;
    int qg = tile * QPB + q;
    float cx = coords[qg], cy = coords[HW + qg];
    float fx = cx - floorf(cx), fy = cy - floorf(cy);
    const __bf16* Wn = s_win + q * 100;
    float v00 = (float)Wn[iy * 10 + ix];
    float v01 = (float)Wn[iy * 10 + ix + 1];
    float v10 = (float)Wn[(iy + 1) * 10 + ix];
    float v11 = (float)Wn[(iy + 1) * 10 + ix + 1];
    out[o * HW + qg] = (1.f - fy) * ((1.f - fx) * v00 + fx * v01) +
                       fy * ((1.f - fx) * v10 + fx * v11);
  }
  pool_level(s_l1, s_l2, 16, 32, tid);
  __syncthreads();

  // B2: sample level-1; pool L2 -> L3.
  sample_level(s_l1, 1, 32, 64, coords, tile, out, tid);
  pool_level(s_l2, s_l3, 8, 16, tid);
  __syncthreads();

  // B3: sample level-2 and level-3.
  sample_level(s_l2, 2, 16, 32, coords, tile, out, tid);
  sample_level(s_l3, 3, 8, 16, coords, tile, out, tid);
}

extern "C" void kernel_launch(void* const* d_in, const int* in_sizes, int n_in,
                              void* d_out, int out_size, void* d_ws, size_t ws_size,
                              hipStream_t stream) {
  (void)in_sizes; (void)n_in; (void)out_size; (void)ws_size;
  const float* f1     = (const float*)d_in[0];   // [1,256,64,128]
  const float* f2     = (const float*)d_in[1];   // [1,256,64,128]
  const float* coords = (const float*)d_in[2];   // [1,2,64,128] (x then y)
  __bf16* At = (__bf16*)d_ws;                    // [8192,256] bf16 (pre-scaled)
  __bf16* Bt = At + HW * CH;                     // [8192,256] bf16
  float* out = (float*)d_out;                    // [1,324,64,128]

  const int prep_elems = 2 * HW * CH;
  corr_prep_kernel<<<prep_elems / TPB, TPB, 0, stream>>>(f1, f2, At, Bt);
  corr_main_kernel<<<NTILES, TPB, 0, stream>>>(At, Bt, coords, out);
}
